// VQVAE_10608569221272
// MI455X (gfx1250) — matrix operand order repaired and checked by reading, not compiled, not run on hardware
//
#include <hip/hip_runtime.h>
#include <hip/hip_bf16.h>
#include <stdint.h>

typedef __attribute__((ext_vector_type(16))) __bf16 v16bf;
typedef __attribute__((ext_vector_type(8)))  __bf16 v8bf;
typedef __attribute__((ext_vector_type(4)))  __bf16 bf16x4;
typedef __attribute__((ext_vector_type(8)))  float  v8f;
typedef __attribute__((ext_vector_type(4)))  float  f32x4;
typedef __attribute__((ext_vector_type(4)))  unsigned int u32x4;
typedef __attribute__((ext_vector_type(8)))  unsigned int u32x8;

#define BM 128
#define BN 128
#define BK 32
#define LDT 40   // LDS stride (bf16): 80B = 64B tile row + 16B TDM pad -> conflict-free b128 reads

enum { ACT_NONE = 0, ACT_RELU = 1, ACT_SIGMOID = 2 };

// ---- Tensor Data Mover: async 2D bf16 tile (BKxBM elems, row stride = stride_elems) -> LDS ----
// D# group0/group1 per CDNA5 ISA. 2D => groups 2/3 omitted.
// pad_enable=1, pad_interval=3 (16 DWORDs = one 64B tile row), pad_amount=3 (4 DWORDs = 16B)
// reproduces the LDT=40 padded LDS layout. tensor_dim = remaining extent from tile start;
// weights are zero-padded to tile multiples so every element is in-bounds.
__device__ __forceinline__ void tdm_load_tile_bf16(
    const __bf16* gtile, uint32_t lds_byte,
    uint32_t td0, uint32_t td1, uint32_t stride_elems)
{
    const uint64_t ga = (uint64_t)(uintptr_t)gtile;
    u32x4 g0;
    g0[0] = 1u;                                   // count=1, user descriptor
    g0[1] = lds_byte;                             // lds_addr
    g0[2] = (uint32_t)ga;                         // global_addr[31:0]
    g0[3] = (uint32_t)(ga >> 32) | (2u << 30);    // global_addr[56:32] | type=2 (image)
    u32x8 g1;
    g1[0] = (1u << 16)            // data_size = 1 (2-byte elems)
          | (1u << 20)            // pad_enable
          | (3u << 22)            // pad_interval: 16 DWORDs
          | (3u << 25);           // pad_amount:   4 DWORDs
    g1[1] = (td0 & 0xFFFFu) << 16;                        // tensor_dim0[15:0]
    g1[2] = (td0 >> 16) | ((td1 & 0xFFFFu) << 16);        // dim0 hi | dim1 lo
    g1[3] = (td1 >> 16) | ((uint32_t)BK << 16);           // dim1 hi | tile_dim0=32
    g1[4] = (uint32_t)BM;                                 // tile_dim1=128, tile_dim2=0
    g1[5] = stride_elems;                                 // tensor_dim0_stride[31:0]
    g1[6] = 0u;
    g1[7] = 0u;
    asm volatile("tensor_load_to_lds %0, %1" :: "s"(g0), "s"(g1) : "memory");
}

// C = act(A[M,K] * B + bias[N]); B pre-transposed bf16 [Np][Kp] (zero-padded), row stride Kp.
// AT=float: A converted f32->bf16 through VGPRs. AT=bf16: A tile moved by TDM.
template <typename AT, typename OT, int ACT>
__global__ __launch_bounds__(256) void gemm_wmma(
    const AT* __restrict__ A, const __bf16* __restrict__ Bt,
    const float* __restrict__ bias, OT* __restrict__ C,
    int M, int N, int K, int Kp, int Np)
{
    __shared__ __align__(16) __bf16 As[2][BM * LDT];
    __shared__ __align__(16) __bf16 Bs[2][BN * LDT];

    const int tid  = threadIdx.x;
    const int lane = tid & 31;
    const int wm   = (tid >> 5) & 1;   // 2 waves along M
    const int wn   = tid >> 6;         // 4 waves along N
    const int m0   = blockIdx.x * BM;
    const int n0   = blockIdx.y * BN;
    const uint32_t wave_u = (uint32_t)__builtin_amdgcn_readfirstlane(tid >> 5); // scalar branch

    constexpr bool A_TDM = (sizeof(AT) == 2);

    v8f acc[4][2];
#pragma unroll
    for (int i = 0; i < 4; ++i)
#pragma unroll
        for (int j = 0; j < 2; ++j)
#pragma unroll
            for (int r = 0; r < 8; ++r) acc[i][j][r] = 0.0f;

    const int nk = (K + BK - 1) / BK;

    // ---- manual A stage (f32 -> bf16), only for AT=float ----
    auto stageA = [&](int buf, int k0) {
        const int quad = tid & 7;
        const int r0   = tid >> 3;
#pragma unroll
        for (int it = 0; it < 4; ++it) {
            const int row = r0 + 32 * it;
            const int k   = k0 + quad * 4;
            bf16x4 v;
            v.x = v.y = v.z = v.w = (__bf16)0.0f;
            if (k < K) {   // K multiple of 4 -> whole quad in/out
                const f32x4 f = *(const f32x4*)(A + (size_t)(m0 + row) * K + k);
                v.x = (__bf16)f.x; v.y = (__bf16)f.y;
                v.z = (__bf16)f.z; v.w = (__bf16)f.w;
            }
            *(bf16x4*)&As[buf][row * LDT + quad * 4] = v;
        }
    };

    // ---- issue tile movement for one K-slab ----
    auto issue = [&](int buf, int k0) {
        if (wave_u == 0) {  // one wave issues TDM descriptors (EXEC is ignored by TENSOR ops)
            tdm_load_tile_bf16(Bt + (size_t)n0 * Kp + k0,
                               (uint32_t)(uintptr_t)&Bs[buf][0],
                               (uint32_t)(Kp - k0), (uint32_t)(Np - n0), (uint32_t)Kp);
            if constexpr (A_TDM)
                tdm_load_tile_bf16((const __bf16*)A + (size_t)m0 * K + k0,
                                   (uint32_t)(uintptr_t)&As[buf][0],
                                   (uint32_t)(K - k0), (uint32_t)(M - m0), (uint32_t)K);
        }
        if constexpr (!A_TDM) stageA(buf, k0);
    };

    // ---- per-wave compute: 4x2 WMMA tiles, K-slab of 32 ----
    auto compute = [&](int buf) {
        const int l16   = lane & 15;
        const int lh    = lane >> 4;
        const int khigh = lh * 8;
        v16bf bfr[2];
#pragma unroll
        for (int j = 0; j < 2; ++j) {
            const __bf16* p = &Bs[buf][(wn * 32 + j * 16 + l16) * LDT + lh * 8];
            ((v8bf*)&bfr[j])[0] = *(const v8bf*)p;
            ((v8bf*)&bfr[j])[1] = *(const v8bf*)(p + 16);
        }
#pragma unroll
        for (int i = 0; i < 4; ++i) {
            const __bf16* p = &As[buf][(wm * 64 + i * 16 + l16) * LDT + khigh];
            v16bf afr;
            ((v8bf*)&afr)[0] = *(const v8bf*)p;         // k = khigh..+7
            ((v8bf*)&afr)[1] = *(const v8bf*)(p + 16);  // k = 16+khigh..+7
#pragma unroll
            for (int j = 0; j < 2; ++j) {
                acc[i][j] = __builtin_amdgcn_wmma_f32_16x16x32_bf16(
                    false, afr, false, bfr[j], (short)0, acc[i][j], false, false);
            }
        }
    };

    issue(0, 0);
    __builtin_amdgcn_s_wait_tensorcnt(0);
    __syncthreads();
    int p = 0;
    for (int ks = 0; ks < nk; ++ks) {
        if (ks + 1 < nk) issue(p ^ 1, (ks + 1) * BK);
        compute(p);
        __builtin_amdgcn_s_wait_tensorcnt(0);  // no-op for waves with TENSORcnt==0
        __syncthreads();
        p ^= 1;
    }

    // ---- epilogue: bias + activation (C/D layout: n = lane%16, m = r + 8*(lane/16)) ----
    const int l16 = lane & 15;
    const int lh  = lane >> 4;
#pragma unroll
    for (int j = 0; j < 2; ++j) {
        const int n = n0 + wn * 32 + j * 16 + l16;
        if (n >= N) continue;
        const float bv = bias[n];
#pragma unroll
        for (int i = 0; i < 4; ++i) {
            const int mb = m0 + wm * 64 + i * 16 + 8 * lh;
#pragma unroll
            for (int r = 0; r < 8; ++r) {
                float v = acc[i][j][r] + bv;
                if constexpr (ACT == ACT_RELU)    v = v > 0.0f ? v : 0.0f;
                if constexpr (ACT == ACT_SIGMOID) v = 1.0f / (1.0f + __expf(-v));
                C[(size_t)(mb + r) * N + n] = (OT)v;
            }
        }
    }
}

// W [K][N] f32 -> Wt [Np][Kp] bf16, transposed + zero-padded (output-coalesced).
__global__ __launch_bounds__(256) void transpose_w(
    const float* __restrict__ W, __bf16* __restrict__ Wt, int K, int N, int Kp, int Np)
{
    const int idx = blockIdx.x * 256 + threadIdx.x;
    if (idx >= Kp * Np) return;
    const int n = idx / Kp, k = idx % Kp;
    const float v = (n < N && k < K) ? W[(size_t)k * N + n] : 0.0f;
    Wt[idx] = (__bf16)v;
}

// One row per thread: argmin_k ||z - c_k||^2, K=16 codes, D=64.
__global__ __launch_bounds__(256) void quantize_kernel(
    const float* __restrict__ z_e, const float* __restrict__ codebook,
    float* __restrict__ z_q, int* __restrict__ inds,
    __bf16* __restrict__ zq_bf, int Brows)
{
    __shared__ float cb[16 * 64];
    for (int i = threadIdx.x; i < 16 * 64; i += 256) cb[i] = codebook[i];
    __syncthreads();

    const int row = blockIdx.x * 256 + threadIdx.x;
    if (row >= Brows) return;

    float z[64];
    const float* zp = z_e + (size_t)row * 64;
#pragma unroll
    for (int i = 0; i < 16; ++i) *(f32x4*)&z[i * 4] = *(const f32x4*)(zp + i * 4);

    int best = 0;
    float bd = 3.4e38f;
#pragma unroll
    for (int k = 0; k < 16; ++k) {
        const float* c = &cb[k * 64];
        float d = 0.0f;
#pragma unroll
        for (int i = 0; i < 64; ++i) { const float t = z[i] - c[i]; d = fmaf(t, t, d); }
        if (d < bd) { bd = d; best = k; }   // strict <: first index, matches argmin
    }

    inds[row] = best;
    const float* c = &cb[best * 64];
    float*  q  = z_q  + (size_t)row * 64;
    __bf16* qb = zq_bf + (size_t)row * 64;
#pragma unroll
    for (int i = 0; i < 64; ++i) { q[i] = c[i]; qb[i] = (__bf16)c[i]; }
}

extern "C" void kernel_launch(void* const* d_in, const int* in_sizes, int n_in,
                              void* d_out, int out_size, void* d_ws, size_t ws_size,
                              hipStream_t stream)
{
    (void)in_sizes; (void)n_in; (void)out_size; (void)ws_size;

    const float* x      = (const float*)d_in[0];
    const float* enc_w1 = (const float*)d_in[1];
    const float* enc_b1 = (const float*)d_in[2];
    const float* enc_w2 = (const float*)d_in[3];
    const float* enc_b2 = (const float*)d_in[4];
    const float* cbk    = (const float*)d_in[5];
    const float* dec_w1 = (const float*)d_in[6];
    const float* dec_b1 = (const float*)d_in[7];
    const float* dec_w2 = (const float*)d_in[8];
    const float* dec_b2 = (const float*)d_in[9];

    const int B = 32768, D = 64, H = 512, P = 784;
    const int Pp = 800;   // 784 padded to BK multiple (K of enc1)
    const int Dp = 128;   // 64 padded to BN multiple  (N of enc2)
    const int Pn = 896;   // 784 padded to BN multiple (N of dec2)

    // Outputs concatenated: x_hat [B*784], z_e [B*64], z_q [B*64], inds [B]
    float* out   = (float*)d_out;
    float* x_hat = out;
    float* z_e   = out + (size_t)B * P;
    float* z_q   = z_e + (size_t)B * D;
    int*   inds  = (int*)(z_q + (size_t)B * D);

    // Workspace: bf16 intermediates + pre-transposed bf16 weights (~73 MB)
    __bf16* h_bf  = (__bf16*)d_ws;                  // [B, 512]
    __bf16* h2_bf = h_bf  + (size_t)B * H;          // [B, 512]
    __bf16* zq_bf = h2_bf + (size_t)B * H;          // [B, 64]
    __bf16* w1t   = zq_bf + (size_t)B * D;          // [512][800]
    __bf16* w2t   = w1t + (size_t)H * Pp;           // [128][512]
    __bf16* d1t   = w2t + (size_t)Dp * H;           // [512][64]
    __bf16* d2t   = d1t + (size_t)H * D;            // [896][512]

    const dim3 blk(256);
    auto cdiv = [](int a, int b) { return (a + b - 1) / b; };

    // One-time weight transpose/convert (f32 [K][N] -> bf16 [Np][Kp], zero-padded)
    transpose_w<<<cdiv(H * Pp, 256), blk, 0, stream>>>(enc_w1, w1t, P, H, Pp, H);
    transpose_w<<<cdiv(Dp * H, 256), blk, 0, stream>>>(enc_w2, w2t, H, D, H, Dp);
    transpose_w<<<cdiv(H * D, 256),  blk, 0, stream>>>(dec_w1, d1t, D, H, D, H);
    transpose_w<<<cdiv(Pn * H, 256), blk, 0, stream>>>(dec_w2, d2t, H, P, H, Pn);

    // enc1: relu(x @ enc_w1 + b1) -> h (bf16)
    gemm_wmma<float, __bf16, ACT_RELU>
        <<<dim3(B / BM, cdiv(H, BN)), blk, 0, stream>>>(x, w1t, enc_b1, h_bf, B, H, P, Pp, H);
    // enc2: h @ enc_w2 + b2 -> z_e (f32)
    gemm_wmma<__bf16, float, ACT_NONE>
        <<<dim3(B / BM, cdiv(D, BN)), blk, 0, stream>>>(h_bf, w2t, enc_b2, z_e, B, D, H, H, Dp);
    // quantize
    quantize_kernel<<<dim3(B / 256), blk, 0, stream>>>(z_e, cbk, z_q, inds, zq_bf, B);
    // dec1: relu(z_q @ dec_w1 + b1) -> h2 (bf16)
    gemm_wmma<__bf16, __bf16, ACT_RELU>
        <<<dim3(B / BM, cdiv(H, BN)), blk, 0, stream>>>(zq_bf, d1t, dec_b1, h2_bf, B, H, D, D, H);
    // dec2: sigmoid(h2 @ dec_w2 + b2) -> x_hat (f32)
    gemm_wmma<__bf16, float, ACT_SIGMOID>
        <<<dim3(B / BM, cdiv(P, BN)), blk, 0, stream>>>(h2_bf, d2t, dec_b2, x_hat, B, P, H, H, Pn);
}
